// SE3Transformer_18631568130052
// MI455X (gfx1250) — matrix operand order repaired and checked by reading, not compiled, not run on hardware
//
#include <hip/hip_runtime.h>
#include <math.h>

#define N_NODES 8192
#define N_EDGES 262144

typedef __attribute__((ext_vector_type(16))) _Float16 v16h;
typedef __attribute__((ext_vector_type(8)))  _Float16 h8;
typedef __attribute__((ext_vector_type(8)))  float    v8f;
typedef unsigned int u32x4 __attribute__((ext_vector_type(4)));
typedef unsigned int u32x8 __attribute__((ext_vector_type(8)));

__device__ __forceinline__ v8f wmma_f16(v16h a, v16h b, v8f c) {
  // D = A(16x32 f16) x B(32x16 f16) + C(16x16 f32)
  return __builtin_amdgcn_wmma_f32_16x16x32_f16(false, a, false, b, (short)0, c, false, false);
}

// A fragment (16x32, f16) from LDS, row stride ld (elements).
// Wave32 layout: lane m=L%16 holds row m; lanes 0-15: K={0..7,16..23}, lanes 16-31: K={8..15,24..31}
__device__ __forceinline__ v16h a_frag(const _Float16* S, int ld, int lane) {
  int mm = lane & 15, half = lane >> 4;
  v16h a;
#pragma unroll
  for (int e = 0; e < 16; ++e) {
    int k = (e & 7) + 8 * (2 * (e >> 3) + half);
    a[e] = S[mm * ld + k];
  }
  return a;
}

// B fragment (32x16, f16) from f32 row-major weight W[k][col0+n] (global or LDS), row stride ld.
// Rows k >= kmax zero-padded; values scaled by sc.
// Layout: lane n=L%16 holds column n; lanes 0-15: K={0..7,16..23}, lanes 16-31: K={8..15,24..31}
__device__ __forceinline__ v16h b_frag(const float* W, int ld, int col0, int kmax, float sc, int lane) {
  int n = lane & 15, half = lane >> 4;
  v16h b;
#pragma unroll
  for (int e = 0; e < 16; ++e) {
    int k = (e & 7) + 8 * (2 * (e >> 3) + half);
    float v = (k < kmax) ? W[k * ld + col0 + n] * sc : 0.0f;
    b[e] = (_Float16)v;
  }
  return b;
}

// ---- CDNA5 async / TDM helpers (inline asm per CDNA5 ISA) ----
__device__ __forceinline__ void async_ld_b128(unsigned lds_off, const void* gaddr) {
  // memory -> LDS, tracked by ASYNCcnt; per-lane addresses
  asm volatile("global_load_async_to_lds_b128 %0, %1, off"
               :: "v"(lds_off), "v"(gaddr) : "memory");
}
__device__ __forceinline__ void wait_asynccnt0() {
  asm volatile("s_wait_asynccnt 0x0" ::: "memory");
}

// TDM: 2D tile load, data_size=4B, tile == tensor (dim0 x dim1), row stride dim0.
__device__ __forceinline__ void tdm_load_2d(const void* gptr, unsigned lds_off,
                                            unsigned dim0, unsigned dim1) {
  unsigned long long ga = (unsigned long long)(size_t)gptr;
  u32x4 g0;
  g0[0] = 1u;                                              // count=1, user D#
  g0[1] = lds_off;                                         // lds_addr (bytes)
  g0[2] = (unsigned)(ga & 0xFFFFFFFFu);                    // global_addr[31:0]
  g0[3] = (unsigned)((ga >> 32) & 0x01FFFFFFu) | (2u << 30); // addr[56:32] | type=2
  u32x8 g1;
  g1[0] = 2u << 16;                                        // data_size: 4 bytes
  g1[1] = (dim0 & 0xFFFFu) << 16;                          // tensor_dim0[15:0]
  g1[2] = (dim0 >> 16) | ((dim1 & 0xFFFFu) << 16);         // dim0[31:16] | dim1[15:0]
  g1[3] = (dim1 >> 16) | ((dim0 & 0xFFFFu) << 16);         // dim1[31:16] | tile_dim0
  g1[4] = dim1 & 0xFFFFu;                                  // tile_dim1 (tile_dim2=0)
  g1[5] = dim0;                                            // tensor_dim0_stride[31:0]
  g1[6] = 0u;
  g1[7] = 0u;
  asm volatile("tensor_load_to_lds %0, %1" :: "s"(g0), "s"(g1) : "memory");
}

// ---------------------------------------------------------------------------
// Kernel 1: node-level work (one wave = 16 nodes).
//   qd = ((s@Wq)/sqrt32)@Wd  -> ws (planar: cols 0..31 scalar, 32+32c vector ch c)
//   d_out init = skip connection ; z = 0
// ---------------------------------------------------------------------------
__global__ __launch_bounds__(32)
void node_kernel(const float* __restrict__ node_feats, const float* __restrict__ node_attrs,
                 const float* __restrict__ Wq0, const float* __restrict__ Wq1,
                 const float* __restrict__ Wd0, const float* __restrict__ Wd1,
                 const float* __restrict__ Ws0, const float* __restrict__ Ws1,
                 float* __restrict__ qd, float* __restrict__ z, float* __restrict__ out) {
  __shared__ float featf[16 * 128];
  __shared__ float attrf[16 * 16];
  __shared__ _Float16 a16[16 * 72];

  int lane = threadIdx.x;
  int m = lane & 15, half = lane >> 4;
  int base = blockIdx.x * 16;

  if (half == 0) z[base + m] = 0.0f;

#pragma unroll 8
  for (int t = 0; t < 64; ++t) {
    int idx = t * 32 + lane;
    featf[idx] = node_feats[(size_t)(base + (idx >> 7)) * 128 + (idx & 127)];
  }
#pragma unroll
  for (int t = 0; t < 8; ++t) {
    int idx = t * 32 + lane;
    attrf[idx] = node_attrs[(size_t)(base + (idx >> 4)) * 16 + (idx & 15)];
  }
  __syncthreads();

  const float qs = 0.17677670f;   // 1/sqrt(MUL)
#pragma unroll 1
  for (int comp = 0; comp < 4; ++comp) {
#pragma unroll
    for (int t = 0; t < 16; ++t) {
      int idx = t * 32 + lane; int row = idx >> 5; int u = idx & 31;
      float v = (comp == 0) ? featf[row * 128 + u]
                            : featf[row * 128 + 32 + u * 3 + (comp - 1)];
      a16[row * 72 + u] = (_Float16)(v * qs);
    }
    __syncthreads();
    v16h a = a_frag(a16, 72, lane);
    const float* Wq = (comp == 0) ? Wq0 : Wq1;
    v8f qc[2];
#pragma unroll
    for (int nt = 0; nt < 2; ++nt) {
      v8f c = {};
      qc[nt] = wmma_f16(a, b_frag(Wq, 32, nt * 16, 32, 1.0f, lane), c);
    }
    __syncthreads();
#pragma unroll
    for (int nt = 0; nt < 2; ++nt)
#pragma unroll
      for (int r = 0; r < 8; ++r)
        a16[(r + 8 * half) * 72 + nt * 16 + m] = (_Float16)qc[nt][r];
    __syncthreads();
    v16h a2 = a_frag(a16, 72, lane);
    const float* Wd = (comp == 0) ? Wd0 : Wd1;
#pragma unroll
    for (int nt = 0; nt < 2; ++nt) {
      v8f c = {};
      c = wmma_f16(a2, b_frag(Wd, 32, nt * 16, 32, 1.0f, lane), c);
#pragma unroll
      for (int r = 0; r < 8; ++r)
        qd[(size_t)(base + r + 8 * half) * 128 + comp * 32 + nt * 16 + m] = c[r];
    }
    __syncthreads();
  }

  // skip connection
  const float ss = 0.04419417f;   // 1/sqrt(MUL*ATTR)
#pragma unroll 1
  for (int comp = 0; comp < 4; ++comp) {
    v8f acc0 = {}, acc1 = {};
#pragma unroll 1
    for (int ai = 0; ai < 16; ++ai) {
#pragma unroll
      for (int t = 0; t < 16; ++t) {
        int idx = t * 32 + lane; int row = idx >> 5; int u = idx & 31;
        float v = (comp == 0) ? featf[row * 128 + u]
                              : featf[row * 128 + 32 + u * 3 + (comp - 1)];
        a16[row * 72 + u] = (_Float16)(v * attrf[row * 16 + ai]);
      }
      __syncthreads();
      v16h a = a_frag(a16, 72, lane);
      const float* Ws = ((comp == 0) ? Ws0 : Ws1) + ai * 32;   // Ws[u,a,w]: row stride 512
      acc0 = wmma_f16(a, b_frag(Ws, 512, 0, 32, 1.0f, lane), acc0);
      acc1 = wmma_f16(a, b_frag(Ws, 512, 16, 32, 1.0f, lane), acc1);
      __syncthreads();
    }
#pragma unroll
    for (int nt = 0; nt < 2; ++nt) {
      v8f acc = nt ? acc1 : acc0;
#pragma unroll
      for (int r = 0; r < 8; ++r) {
        int w = nt * 16 + m;
        int col = (comp == 0) ? w : 32 + w * 3 + (comp - 1);
        out[(size_t)(base + r + 8 * half) * 128 + col] = acc[r] * ss;
      }
    }
  }
}

// ---------------------------------------------------------------------------
// wf = ssp(emb@W1 / sqrt(16)) @ W2 / sqrt(64)   (W2 read from LDS, TDM-staged)
// ---------------------------------------------------------------------------
__device__ __forceinline__ void compute_w(const float* __restrict__ emb,
                                          const float* W1, const float* W2lds,
                                          int e0, int lane, _Float16* a16, float* wf) {
  int m = lane & 15, half = lane >> 4;
#pragma unroll
  for (int t = 0; t < 16; ++t) {
    int idx = t * 32 + lane; int row = idx >> 5; int col = idx & 31;
    float v = (col < 16) ? emb[(size_t)(e0 + row) * 16 + col] * 0.25f : 0.0f;
    a16[row * 72 + col] = (_Float16)v;
  }
  __syncthreads();
  v16h a = a_frag(a16, 72, lane);
  v8f hc[4];
#pragma unroll
  for (int j = 0; j < 4; ++j) {
    v8f c = {};
    hc[j] = wmma_f16(a, b_frag(W1, 64, j * 16, 16, 1.0f, lane), c);   // K padded 16->32
  }
  __syncthreads();
#pragma unroll
  for (int j = 0; j < 4; ++j)
#pragma unroll
    for (int r = 0; r < 8; ++r) {
      float x = hc[j][r];
      float y = (x > 20.0f) ? x : log1pf(expf(x));   // softplus
      y -= 0.69314718f;                               // - log(2)
      a16[(r + 8 * half) * 72 + j * 16 + m] = (_Float16)(y * 0.125f);
    }
  __syncthreads();
#pragma unroll
  for (int jj = 0; jj < 8; ++jj) {
    v8f c = {};
#pragma unroll
    for (int kk = 0; kk < 2; ++kk) {
      v16h aa = a_frag(a16 + kk * 32, 72, lane);
      c = wmma_f16(aa, b_frag(W2lds + kk * 32 * 128, 128, jj * 16, 32, 1.0f, lane), c);
    }
#pragma unroll
    for (int r = 0; r < 8; ++r)
      wf[(r + 8 * half) * 128 + jj * 16 + m] = c[r];
  }
  __syncthreads();
}

// ---------------------------------------------------------------------------
// Kernel 2 (edge pass 1): 64 threads = 2 waves; each wave does 8 tiles of 16 edges.
// Wk2/Wv2 staged to LDS by TDM once per block; node rows gathered by async LDS DMA.
// ---------------------------------------------------------------------------
__global__ __launch_bounds__(64)
void edge_pass1(const float* __restrict__ node_feats, const float* __restrict__ emb,
                const float* __restrict__ edge_attrs, const float* __restrict__ positions,
                const float* __restrict__ Wk1, const float* __restrict__ Wk2,
                const float* __restrict__ Wv1, const float* __restrict__ Wv2,
                const float* __restrict__ Wlk0, const float* __restrict__ Wlk1,
                const float* __restrict__ Wlv0, const float* __restrict__ Wlv1,
                const int* __restrict__ src, const int* __restrict__ dstv,
                const float* __restrict__ qd, float* __restrict__ z,
                float* __restrict__ expv, _Float16* __restrict__ v16g) {
  __shared__ float wk2s[64 * 128];             // 32KB, TDM target
  __shared__ float wv2s[64 * 128];             // 32KB, TDM target
  __shared__ int   sidxs[2][16], didxs[2][16];
  __shared__ float yats[2][64];
  __shared__ float xsfs[2][16 * 128];          // async-DMA gather target
  __shared__ float wfs[2][16 * 128];
  __shared__ float kfs[2][16 * 128];
  __shared__ _Float16 a16s[2][16 * 72];
  __shared__ float reds[2][32];

  int tid = threadIdx.x;
  int wave = tid >> 5, lane = tid & 31;
  int m = lane & 15, half = lane >> 4;

  int* sidx = sidxs[wave]; int* didx = didxs[wave];
  float* yat = yats[wave]; float* xsf = xsfs[wave];
  float* wf = wfs[wave];   float* kf = kfs[wave];
  _Float16* a16 = a16s[wave]; float* red = reds[wave];

  // ---- TDM: stage the two big weight matrices once per block (wave 0 issues) ----
  if (tid < 32) {
    tdm_load_2d(Wk2, (unsigned)(size_t)(void*)wk2s, 128u, 64u);
    tdm_load_2d(Wv2, (unsigned)(size_t)(void*)wv2s, 128u, 64u);
    __builtin_amdgcn_s_wait_tensorcnt(0);
  }
  __syncthreads();

  unsigned xsf_off = (unsigned)(size_t)(void*)xsf;
  const float inv_s3 = 0.57735027f;   // 1/sqrt(3)

#pragma unroll 1
  for (int it = 0; it < 8; ++it) {
    int e0 = (blockIdx.x * 16 + it * 2 + wave) * 16;
    if (half == 0) {
      int s = src[e0 + m], dd = dstv[e0 + m];
      sidx[m] = s; didx[m] = dd;
      __builtin_prefetch(&qd[(size_t)dd * 128], 0, 1);   // global_prefetch_b8
    }
#pragma unroll
    for (int t = 0; t < 2; ++t) {
      int idx = t * 32 + lane;
      yat[idx] = edge_attrs[(size_t)(e0 + (idx >> 2)) * 4 + (idx & 3)];
    }
    __syncthreads();
    // ---- async gather: 16 node rows (512B each) -> LDS, 16B per lane-op ----
#pragma unroll
    for (int t = 0; t < 16; ++t) {
      int c = t * 32 + lane;             // 512 x 16B chunks
      int row = c >> 5, c16 = c & 31;
      const char* g = (const char*)node_feats + (size_t)sidx[row] * 512 + (size_t)c16 * 16;
      async_ld_b128(xsf_off + (unsigned)c * 16u, g);
    }
    wait_asynccnt0();
    __syncthreads();

#pragma unroll 1
    for (int pass = 0; pass < 2; ++pass) {
      compute_w(emb, pass ? Wv1 : Wk1, pass ? wv2s : wk2s, e0, lane, a16, wf);
      const float* Wl0 = pass ? Wlv0 : Wlk0;
      const float* Wl1 = pass ? Wlv1 : Wlk1;
#pragma unroll 1
      for (int comp = 0; comp < 4; ++comp) {
        // uvu tensor product -> A tile (16x64), 1/sqrt(64) lin scale folded in
#pragma unroll 4
        for (int t = 0; t < 32; ++t) {
          int idx = t * 32 + lane; int row = idx >> 6; int u = idx & 63;
          float y0 = yat[row * 4 + 0];
          float v;
          if (comp == 0) {
            if (u < 32) {
              v = wf[row * 128 + u] * xsf[row * 128 + u] * y0;                 // w0*xs0*y0
            } else {
              int uu = u - 32;
              float dp = xsf[row * 128 + 32 + uu * 3 + 0] * yat[row * 4 + 1]
                       + xsf[row * 128 + 32 + uu * 3 + 1] * yat[row * 4 + 2]
                       + xsf[row * 128 + 32 + uu * 3 + 2] * yat[row * 4 + 3];
              v = wf[row * 128 + 96 + uu] * dp * inv_s3;                      // w3*<xs1,y1>/sqrt3
            }
          } else {
            int c = comp - 1;
            if (u < 32) {
              v = wf[row * 128 + 32 + u] * xsf[row * 128 + u] * yat[row * 4 + 1 + c]; // w1*xs0*y1c
            } else {
              int uu = u - 32;
              v = wf[row * 128 + 64 + uu] * xsf[row * 128 + 32 + uu * 3 + c] * y0;    // w2*xs1c*y0
            }
          }
          a16[row * 72 + u] = (_Float16)(v * 0.125f);
        }
        __syncthreads();
        const float* Wl = (comp == 0) ? Wl0 : Wl1;
#pragma unroll
        for (int nt = 0; nt < 2; ++nt) {
          v8f c = {};
#pragma unroll
          for (int kk = 0; kk < 2; ++kk) {
            v16h aa = a_frag(a16 + kk * 32, 72, lane);
            c = wmma_f16(aa, b_frag(Wl + kk * 32 * 32, 32, nt * 16, 32, 1.0f, lane), c);
          }
          if (pass == 0) {
#pragma unroll
            for (int r = 0; r < 8; ++r)
              kf[(r + 8 * half) * 128 + comp * 32 + nt * 16 + m] = c[r];
          } else {
#pragma unroll
            for (int r = 0; r < 8; ++r) {
              int vv = nt * 16 + m;
              int col = (comp == 0) ? vv : 32 + vv * 3 + (comp - 1);   // output layout
              v16g[(size_t)(e0 + r + 8 * half) * 128 + col] = (_Float16)c[r];
            }
          }
        }
        __syncthreads();
      }
      if (pass == 0) {
        // d = <qd[dst], k> (channels weighted 1/sqrt3), /sqrt(2*MUL*MUL)
        int dd = didx[m];
        float p = 0.0f;
#pragma unroll 8
        for (int jj = 0; jj < 64; ++jj) {
          int j = half * 64 + jj;
          float wgt = (j < 32) ? 1.0f : inv_s3;
          p += qd[(size_t)dd * 128 + j] * kf[m * 128 + j] * wgt;
        }
        red[lane] = p;
        __syncthreads();
        if (half == 0) {
          float d = (red[m] + red[m + 16]) * 0.02209709f;  // 1/sqrt(2048)
          int s = sidx[m];
          float px = positions[s * 3 + 0], py = positions[s * 3 + 1], pz = positions[s * 3 + 2];
          float dx = px - px, dy = py - py, dz = pz - pz;  // reference: src for both endpoints
          float el = sqrtf(dx * dx + dy * dy + dz * dz + 1e-12f);
          float xcut = 10.0f * (1.0f - el * 0.2f);
          float cut = (xcut > 0.0f) ? expf(-1.0f / fmaxf(xcut, 1e-6f)) : 0.0f;
          float evv = cut * expf(d);
          expv[e0 + m] = evv;
          atomicAdd(&z[dd], evv);
        }
        __syncthreads();
      }
    }
  }
}

// ---------------------------------------------------------------------------
// Kernel 3 (edge pass 2): msg = sqrt(relu(expv/z[dst])) * v ; scatter-add (8 cols/thread).
// ---------------------------------------------------------------------------
__global__ __launch_bounds__(256)
void edge_pass2(const int* __restrict__ dstv, const float* __restrict__ expv,
                const float* __restrict__ z, const _Float16* __restrict__ v16g,
                float* __restrict__ out) {
  unsigned idx8 = blockIdx.x * 256u + threadIdx.x;   // E*16 threads, 8 cols each
  unsigned e = idx8 >> 4, jb = (idx8 & 15u) * 8u;
  int dd = dstv[e];
  float zz = z[dd];
  if (zz == 0.0f) zz = 1.0f;
  float a = sqrtf(fmaxf(expv[e] / zz, 0.0f));
  h8 vv = ((const h8*)v16g)[idx8];
#pragma unroll
  for (int j = 0; j < 8; ++j)
    atomicAdd(&out[(size_t)dd * 128 + jb + j], a * (float)vv[j]);
}

// ---------------------------------------------------------------------------
extern "C" void kernel_launch(void* const* d_in, const int* in_sizes, int n_in,
                              void* d_out, int out_size, void* d_ws, size_t ws_size,
                              hipStream_t stream) {
  const float* node_feats = (const float*)d_in[0];
  const float* node_attrs = (const float*)d_in[1];
  const float* emb        = (const float*)d_in[2];
  const float* edge_attrs = (const float*)d_in[3];
  const float* positions  = (const float*)d_in[4];
  const float* Wq0  = (const float*)d_in[5];
  const float* Wq1  = (const float*)d_in[6];
  const float* Wk1  = (const float*)d_in[7];
  const float* Wk2  = (const float*)d_in[8];
  const float* Wv1  = (const float*)d_in[9];
  const float* Wv2  = (const float*)d_in[10];
  const float* Wlk0 = (const float*)d_in[11];
  const float* Wlk1 = (const float*)d_in[12];
  const float* Wlv0 = (const float*)d_in[13];
  const float* Wlv1 = (const float*)d_in[14];
  const float* Wd0  = (const float*)d_in[15];
  const float* Wd1  = (const float*)d_in[16];
  const float* Ws0  = (const float*)d_in[17];
  const float* Ws1  = (const float*)d_in[18];
  const int*   eidx = (const int*)d_in[19];
  const int* src  = eidx;
  const int* dstv = eidx + N_EDGES;
  float* out = (float*)d_out;

  // workspace (~69 MB): qd (N x 128 f32), z (N f32), expv (E f32), v (E x 128 f16)
  char* w = (char*)d_ws;
  float* qd = (float*)w;            w += (size_t)N_NODES * 128 * sizeof(float);
  float* z  = (float*)w;            w += (size_t)N_NODES * sizeof(float);
  float* ev = (float*)w;            w += (size_t)N_EDGES * sizeof(float);
  _Float16* v16 = (_Float16*)w;

  node_kernel<<<N_NODES / 16, 32, 0, stream>>>(node_feats, node_attrs, Wq0, Wq1, Wd0, Wd1,
                                               Ws0, Ws1, qd, z, out);
  edge_pass1<<<N_EDGES / 256, 64, 0, stream>>>(node_feats, emb, edge_attrs, positions,
                                               Wk1, Wk2, Wv1, Wv2, Wlk0, Wlk1, Wlv0, Wlv1,
                                               src, dstv, qd, z, ev, v16);
  edge_pass2<<<(N_EDGES * 16) / 256, 256, 0, stream>>>(dstv, ev, z, v16, out);
}
